// ContrastiveLossOptimized_42846593744919
// MI455X (gfx1250) — matrix operand order repaired and checked by reading, not compiled, not run on hardware
//
#include <hip/hip_runtime.h>
#include <hip/hip_bf16.h>

// CDNA5 / gfx1250, wave32. Contrastive cosine loss:
//   normalize rows -> Xn (f16), C = Xn * Xn^T via v_wmma_f32_16x16x32_f16,
//   elementwise loss consumed in registers, deterministic two-stage reduction.
// Symmetric: only upper block-triangle computed, off-diagonal blocks weighted 2x.

typedef _Float16 half16 __attribute__((ext_vector_type(16)));
typedef _Float16 half8  __attribute__((ext_vector_type(8)));
typedef float    f32x8  __attribute__((ext_vector_type(8)));

#define NROWS 8192
#define DIM   512
#define NB    64                      // NROWS / 128 blocks per dimension
#define NBLK  (NB * (NB + 1) / 2)     // 2080 upper-triangle blocks
#define EPSN  1e-8f

// ---------------------------------------------------------------------------
// Kernel 1: row L2-normalize f32 -> f16.  One 256-thread block per row.
// ---------------------------------------------------------------------------
__global__ void __launch_bounds__(256)
normalize_kernel(const float* __restrict__ reps, _Float16* __restrict__ xn) {
    __shared__ float red[256];
    const int row = blockIdx.x;
    const int t   = threadIdx.x;
    const float* r = reps + (size_t)row * DIM;
    const float x0 = r[t];
    const float x1 = r[t + 256];
    red[t] = x0 * x0 + x1 * x1;
    __syncthreads();
    for (int s = 128; s > 0; s >>= 1) {
        if (t < s) red[t] += red[t + s];
        __syncthreads();
    }
    const float inv = 1.0f / fmaxf(sqrtf(red[0]), EPSN);
    _Float16* o = xn + (size_t)row * DIM;
    o[t]       = (_Float16)(x0 * inv);
    o[t + 256] = (_Float16)(x1 * inv);
}

// ---------------------------------------------------------------------------
// Kernel 2: tiled WMMA GEMM + fused loss.
//   grid = NBLK blocks of 256 threads (8 wave32 waves).
//   Block (bi,bj), bi<=bj, covers rows [bi*128,+128) x cols [bj*128,+128).
//   Wave w: rows i0 = bi*128 + (w&3)*32 (2 row tiles), cols j0 = bj*128 +
//   (w>>2)*64 (4 col tiles) -> 8 accumulators of v8f.
//   Deterministic: block writes exactly one partial sum, no atomics.
// ---------------------------------------------------------------------------
__global__ void __launch_bounds__(256, 1)
cosloss_kernel(const _Float16* __restrict__ xn,
               const int* __restrict__ labels,
               float* __restrict__ partial) {
    __shared__ float wsum[8];

    // Decode upper-triangular block index (uniform scalar loop).
    int rem = blockIdx.x;
    int bi = 0;
    while (rem >= (NB - bi)) { rem -= (NB - bi); ++bi; }
    const int bj = bi + rem;
    const float weight = (bi == bj) ? 1.0f : 2.0f;

    const int w    = threadIdx.x >> 5;
    const int lane = threadIdx.x & 31;
    const int lh   = lane & 15;   // position within half-wave
    const int hi   = lane >> 4;   // which half-wave

    const int i0 = bi * 128 + (w & 3) * 32;   // this wave's first row
    const int j0 = bj * 128 + (w >> 2) * 64;  // this wave's first col

    // ISA 7.12.2 wave32 fragment layouts (16-bit):
    //  A 16x32: lane(row=lh) holds K = [hi*8, +8) then [hi*8+16, +8)
    //  B 32x16: lane(col=lh) holds K = [hi*8, +8) then [hi*8+16, +8)   (col of B == row of Xn)
    const int kbA = hi * 8;
    const int kbB = hi * 8;
    const _Float16* arow = xn + (size_t)(i0 + lh) * DIM;
    const _Float16* brow = xn + (size_t)(j0 + lh) * DIM;

    f32x8 acc[2][4] = {};

    for (int k0 = 0; k0 < DIM; k0 += 32) {
        half16 a[2], b[4];
#pragma unroll
        for (int m = 0; m < 2; ++m) {
            const _Float16* p = arow + (size_t)(m * 16) * DIM + k0 + kbA;
            half8 lo  = *(const half8*)(p);
            half8 hi8 = *(const half8*)(p + 16);
            a[m] = __builtin_shufflevector(lo, hi8, 0, 1, 2, 3, 4, 5, 6, 7,
                                           8, 9, 10, 11, 12, 13, 14, 15);
        }
#pragma unroll
        for (int n = 0; n < 4; ++n) {
            const _Float16* q = brow + (size_t)(n * 16) * DIM + k0 + kbB;
            half8 blo = *(const half8*)(q);
            half8 bhi = *(const half8*)(q + 16);
            b[n] = __builtin_shufflevector(blo, bhi, 0, 1, 2, 3, 4, 5, 6, 7,
                                           8, 9, 10, 11, 12, 13, 14, 15);
        }
#pragma unroll
        for (int m = 0; m < 2; ++m) {
#pragma unroll
            for (int n = 0; n < 4; ++n) {
                acc[m][n] = __builtin_amdgcn_wmma_f32_16x16x32_f16(
                    /*neg_a=*/false, a[m], /*neg_b=*/false, b[n],
                    /*c_mod=*/(short)0, acc[m][n],
                    /*reuse_a=*/false, /*reuse_b=*/false);
            }
        }
    }

    // C layout: lane L, element r -> row M = r + 8*hi, col N = lh (per tile).
    int labc[4];
#pragma unroll
    for (int n = 0; n < 4; ++n) labc[n] = labels[j0 + n * 16 + lh];

    float lsum = 0.0f;
#pragma unroll
    for (int m = 0; m < 2; ++m) {
        int labr[8];
#pragma unroll
        for (int r = 0; r < 8; ++r) labr[r] = labels[i0 + m * 16 + hi * 8 + r];
#pragma unroll
        for (int n = 0; n < 4; ++n) {
#pragma unroll
            for (int r = 0; r < 8; ++r) {
                const float c = acc[m][n][r];
                const float d = 1.0f - c;               // MARGIN == 1.0
                const bool same = (labr[r] == labc[n]);
                const float v = same ? d : fmaxf(d, 0.0f);
                lsum += v * v;
            }
        }
    }
    lsum *= weight;

    // wave32 butterfly reduction; deterministic block combine through LDS.
#pragma unroll
    for (int off = 16; off > 0; off >>= 1)
        lsum += __shfl_xor(lsum, off);
    if (lane == 0) wsum[w] = lsum;
    __syncthreads();
    if (threadIdx.x == 0) {
        float s = 0.0f;
#pragma unroll
        for (int i = 0; i < 8; ++i) s += wsum[i];
        partial[blockIdx.x] = s;
    }
}

// ---------------------------------------------------------------------------
// Kernel 3: deterministic final reduction of NBLK partials -> mean over N*N.
// ---------------------------------------------------------------------------
__global__ void __launch_bounds__(256)
final_reduce(const float* __restrict__ partial, float* __restrict__ out) {
    __shared__ float red[256];
    float s = 0.0f;
    for (int i = threadIdx.x; i < NBLK; i += 256) s += partial[i];
    red[threadIdx.x] = s;
    __syncthreads();
    for (int st = 128; st > 0; st >>= 1) {
        if (threadIdx.x < st) red[threadIdx.x] += red[threadIdx.x + st];
        __syncthreads();
    }
    if (threadIdx.x == 0)
        out[0] = red[0] * (1.0f / ((float)NROWS * (float)NROWS));
}

// ---------------------------------------------------------------------------
extern "C" void kernel_launch(void* const* d_in, const int* in_sizes, int n_in,
                              void* d_out, int out_size, void* d_ws, size_t ws_size,
                              hipStream_t stream) {
    const float* reps   = (const float*)d_in[0];
    const int*   labels = (const int*)d_in[1];   // int32 on device (JAX x32)
    float*       out    = (float*)d_out;

    _Float16* xn = (_Float16*)d_ws;                                    // 8 MB
    float* partial = (float*)((char*)d_ws +
                              (size_t)NROWS * DIM * sizeof(_Float16)); // 2080 f32

    normalize_kernel<<<NROWS, 256, 0, stream>>>(reps, xn);
    cosloss_kernel<<<NBLK, 256, 0, stream>>>(xn, labels, partial);
    final_reduce<<<1, 256, 0, stream>>>(partial, out);
}
